// AttentionLanguageModel_33036888441136
// MI455X (gfx1250) — matrix operand order repaired and checked by reading, not compiled, not run on hardware
//
#include <hip/hip_runtime.h>
#include <hip/hip_bf16.h>

// ---------------------------------------------------------------------------
// AttentionLanguageModel forward for MI455X (gfx1250, wave32, WMMA bf16)
// ---------------------------------------------------------------------------

typedef __attribute__((ext_vector_type(16))) __bf16        v16bf;
typedef __attribute__((ext_vector_type(8)))  float         v8f;
typedef __attribute__((ext_vector_type(8)))  unsigned int  v8u;

#define V_   32000
#define D_   1024
#define H_   16
#define L_   4
#define B_   2
#define S_   1024
#define HD_  64
#define MTOT (B_ * S_)        // 2048 token rows

// ---- async global->LDS via inline asm (gfx1250 only) ----------------------
// VDST operand = raw LDS byte offset (dynamic-shared base == offset 0),
// VADDR = 64-bit global address, GV mode ("off" saddr). Tracked by ASYNCcnt.
#if defined(__gfx1250__)
#define ASYNC_LDS 1
#else
#define ASYNC_LDS 0
#endif

#if ASYNC_LDS
__device__ __forceinline__ void async_ld16(const void* g, unsigned int lds_off) {
  asm volatile("global_load_async_to_lds_b128 %0, %1, off"
               :: "v"(lds_off), "v"(g) : "memory");
}
__device__ __forceinline__ void async_wait0() {
  asm volatile("s_wait_asynccnt 0x0" ::: "memory");
}
#endif

// ---- scalar helpers -------------------------------------------------------
__device__ __forceinline__ unsigned short f2b(float f) {
  unsigned int u = __float_as_uint(f);
  unsigned int r = u + 0x7FFFu + ((u >> 16) & 1u);   // round-to-nearest-even
  return (unsigned short)(r >> 16);
}

__device__ __forceinline__ v8f zero8() {
  v8f z;
#pragma unroll
  for (int i = 0; i < 8; ++i) z[i] = 0.0f;
  return z;
}

__device__ __forceinline__ v8f wmma_bf16(v16bf a, v16bf b, v8f c) {
  // D = A(16x32 bf16) * B(32x16 bf16) + C(16x16 f32)
  return __builtin_amdgcn_wmma_f32_16x16x32_bf16(
      /*neg_a=*/false, a, /*neg_b=*/false, b,
      /*c_mod=*/(short)0, c, /*reuse_a=*/false, /*reuse_b=*/false);
}

// ---- fragment loaders (layouts per CDNA5 ISA §7.12.2) ----------
// A 16x32 bf16: lane = (K/8 & 1)*16 + M; VGPR p holds K = 16*(p/4)+2*(p%4)+8*(l>>4)+h
__device__ __forceinline__ v16bf load_a_frag(const unsigned short* base,
                                             int row0, int ld, int koff) {
  const int l  = threadIdx.x & 31;
  const int lr = l & 15;
  const int lh = l >> 4;
  const unsigned short* p = base + (row0 + lr) * ld + koff + 8 * lh;
  v8u u;
  u[0] = *(const unsigned int*)(p + 0);
  u[1] = *(const unsigned int*)(p + 2);
  u[2] = *(const unsigned int*)(p + 4);
  u[3] = *(const unsigned int*)(p + 6);
  u[4] = *(const unsigned int*)(p + 16);
  u[5] = *(const unsigned int*)(p + 18);
  u[6] = *(const unsigned int*)(p + 20);
  u[7] = *(const unsigned int*)(p + 22);
  return __builtin_bit_cast(v16bf, u);
}

// B 32x16 bf16 staged as Bt[N][K] in LDS: lane ln = N; VGPR p holds K = 16*(p/4)+2*(p%4)+8*(l>>4)+h
__device__ __forceinline__ v16bf load_b_frag(const unsigned short* base,
                                             int n0, int ld, int koff) {
  const int l  = threadIdx.x & 31;
  const int ln = l & 15;
  const int lh = l >> 4;
  const unsigned short* p = base + (n0 + ln) * ld + koff + 8 * lh;
  v8u u;
#pragma unroll
  for (int i = 0; i < 8; ++i) u[i] = *(const unsigned int*)(p + 2 * i + 8 * (i >> 2));
  return __builtin_bit_cast(v16bf, u);
}

// ---------------------------------------------------------------------------
// Kernel 1: embedding gather + sinusoidal positional encoding
// ---------------------------------------------------------------------------
__global__ __launch_bounds__(256) void embed_pos(
    const int* __restrict__ tokens, const float* __restrict__ emb,
    float* __restrict__ x, unsigned short* __restrict__ xb) {
  const int row = blockIdx.x;            // b*S + s
  const int s   = row & (S_ - 1);
  const int tok = tokens[row];
#pragma unroll
  for (int i = 0; i < 4; ++i) {
    const int d    = threadIdx.x + i * 256;
    const float e  = emb[tok * D_ + d];
    const int half = d >> 1;
    const float dv = __expf(-(float)(2 * half) * (9.210340371976184f / (float)D_));
    const float ang = (float)s * dv;
    const float pe  = (d & 1) ? __cosf(ang) : __sinf(ang);
    const float v   = e + pe;
    x [row * D_ + d] = v;
    xb[row * D_ + d] = f2b(v);
  }
}

// ---------------------------------------------------------------------------
// Kernel 2: WMMA GEMM  C[M,N] = A_bf16[M,K] * W_f32[K,N] (+bias)
//           block tile 128x128, BK=32, 256 threads = 8 waves (4 along M, 2
//           along N), wave tile 32x64 -> 8 v_wmma per k-step.
//           Dynamic LDS: As at byte 0 (async target), Bt follows.
// ---------------------------------------------------------------------------
#define GBM 128
#define GBN 128
#define GBK 32
#define LDA_ 40   // ushorts/row: 80B, 16B-aligned rows for async B128 staging
#define LDB_ 40
#define GEMM_LDS_BYTES ((GBM * LDA_ + GBN * LDB_) * 2)   // 20480

template <int N, int K>
__global__ __launch_bounds__(256) void gemm_xw(
    const unsigned short* __restrict__ A, const float* __restrict__ Bw,
    const float* __restrict__ bias, float* __restrict__ Cf,
    unsigned short* __restrict__ Cb) {
  extern __shared__ unsigned short smem_g[];
  unsigned short* As = smem_g;                 // LDS byte offset 0
  unsigned short* Bt = smem_g + GBM * LDA_;    // transposed [n][k]

  const int tid = threadIdx.x;
  const int wid = tid >> 5;
  const int wm  = wid >> 1;    // 4 waves along M
  const int wn  = wid & 1;     // 2 waves along N
  const int m0  = blockIdx.y * GBM;
  const int n0  = blockIdx.x * GBN;

  v8f acc[2][4];
#pragma unroll
  for (int i = 0; i < 2; ++i)
#pragma unroll
    for (int j = 0; j < 4; ++j) acc[i][j] = zero8();

  for (int k0 = 0; k0 < K; k0 += GBK) {
    // ---- stage A tile: 128x32 bf16 ----
#if ASYNC_LDS
#pragma unroll
    for (int i = 0; i < 2; ++i) {
      const int idx = tid + i * 256;        // 0..511 chunks of 16B
      const int r   = idx >> 2;             // row 0..127
      const int c8  = (idx & 3) << 3;       // col 0,8,16,24
      async_ld16(A + (m0 + r) * K + k0 + c8,
                 (unsigned int)(2 * (r * LDA_ + c8)));   // 80r+16c: 16B aligned
    }
#else
#pragma unroll
    for (int i = 0; i < 8; ++i) {
      const int idx = tid + i * 256;        // 0..2047 dwords
      const int r   = idx >> 4;             // 16 dwords per row
      const int c2  = (idx & 15) << 1;
      const unsigned int d = *(const unsigned int*)(A + (m0 + r) * K + k0 + c2);
      *(unsigned int*)(&As[r * LDA_ + c2]) = d;
    }
#endif
    // ---- stage W tile: 32x128 f32 -> bf16, transposed into Bt[n][k] ----
#pragma unroll
    for (int i = 0; i < 16; ++i) {
      const int idx = tid + i * 256;
      const int r   = idx >> 7;             // k row
      const int c   = idx & 127;            // n col
      const float v = Bw[(k0 + r) * N + n0 + c];
      Bt[c * LDB_ + r] = f2b(v);
    }
    if (k0 + GBK < K)
      __builtin_prefetch(Bw + (k0 + GBK) * N + n0 + (tid & 127), 0, 1);
#if ASYNC_LDS
    async_wait0();
#endif
    __syncthreads();

    v16bf af[2], bf[4];
#pragma unroll
    for (int mf = 0; mf < 2; ++mf)
      af[mf] = load_a_frag(As, wm * 32 + mf * 16, LDA_, 0);
#pragma unroll
    for (int nf = 0; nf < 4; ++nf)
      bf[nf] = load_b_frag(Bt, wn * 64 + nf * 16, LDB_, 0);
#pragma unroll
    for (int mf = 0; mf < 2; ++mf)
#pragma unroll
      for (int nf = 0; nf < 4; ++nf)
        acc[mf][nf] = wmma_bf16(af[mf], bf[nf], acc[mf][nf]);
    __syncthreads();
  }

  // ---- epilogue: C/D layout M = r + 8*(l>>4), N = l&15 ----
  const int l  = tid & 31;
  const int ln = l & 15;
  const int lh = l >> 4;
#pragma unroll
  for (int mf = 0; mf < 2; ++mf)
#pragma unroll
    for (int nf = 0; nf < 4; ++nf)
#pragma unroll
      for (int r = 0; r < 8; ++r) {
        const int gm = m0 + wm * 32 + mf * 16 + r + 8 * lh;
        const int gn = n0 + wn * 64 + nf * 16 + ln;
        float v = acc[mf][nf][r];
        if (bias) v += bias[gn];
        if (Cf) Cf[gm * N + gn] = v;
        if (Cb) Cb[gm * N + gn] = f2b(v);
      }
}

// ---------------------------------------------------------------------------
// Kernel 3: causal flash attention per (b, h, 64-row q block); 4 waves
//           Dynamic LDS layout (byte offsets, all async rows 16B aligned):
//             Qt 0..9215, Kt 9216..18431, Vt 18432..27647, Pt 27648..36863,
//             sc 36864..54271 (f32), alf 54272, linv 54528; total 54784
// ---------------------------------------------------------------------------
#define QT_OFF   0
#define KT_OFF   9216
#define VT_OFF   18432
#define PT_OFF   27648
#define SC_OFF   36864
#define ALF_OFF  54272
#define LINV_OFF 54528
#define ATTN_LDS_BYTES 54784

__global__ __launch_bounds__(128) void attn(
    const unsigned short* __restrict__ Q, const unsigned short* __restrict__ Km,
    const unsigned short* __restrict__ Vm, unsigned short* __restrict__ Ctx) {
  extern __shared__ unsigned char smem_a[];
  unsigned short* Qt = (unsigned short*)(smem_a + QT_OFF);   // [m][hd], ld 72
  unsigned short* Kt = (unsigned short*)(smem_a + KT_OFF);   // [key][hd]
  unsigned short* Vt = (unsigned short*)(smem_a + VT_OFF);   // [hd][key]
  unsigned short* Pt = (unsigned short*)(smem_a + PT_OFF);   // [m][key]
  float*          sc = (float*)(smem_a + SC_OFF);            // [64][68]
  float*         alf = (float*)(smem_a + ALF_OFF);
  float*        linv = (float*)(smem_a + LINV_OFF);

  const int tid = threadIdx.x;
  const int w   = tid >> 5;                // wave 0..3 -> q rows w*16..w*16+15
  const int l   = tid & 31;
  const int ln  = l & 15;
  const int lh  = l >> 4;
  const int qb  = blockIdx.x;              // q block (64 rows)
  const int h   = blockIdx.y;
  const int b   = blockIdx.z;
  const int rowbase = b * S_;
  const int col0    = h * HD_;

  // stage Q tile 64x64 bf16 (rows are 128B -> 8 B128 chunks each)
#if ASYNC_LDS
#pragma unroll
  for (int i = 0; i < 4; ++i) {
    const int idx = tid + i * 128;         // 0..511 chunks
    const int r   = idx >> 3;
    const int c8  = (idx & 7) << 3;
    async_ld16(Q + (rowbase + qb * 64 + r) * D_ + col0 + c8,
               (unsigned int)(QT_OFF + 144 * r + 2 * c8));
  }
#else
#pragma unroll
  for (int i = 0; i < 16; ++i) {
    const int idx = tid + i * 128;
    const int r   = idx >> 5;
    const int c2  = (idx & 31) << 1;
    const unsigned int d =
        *(const unsigned int*)(Q + (rowbase + qb * 64 + r) * D_ + col0 + c2);
    *(unsigned int*)(&Qt[r * 72 + c2]) = d;
  }
#endif

  v8f o[4];
#pragma unroll
  for (int i = 0; i < 4; ++i) o[i] = zero8();
  float m_run = -1e30f, l_run = 0.0f;

  for (int kb = 0; kb <= qb; ++kb) {
    __syncthreads();
    // stage K tile [key][hd]
#if ASYNC_LDS
#pragma unroll
    for (int i = 0; i < 4; ++i) {
      const int idx = tid + i * 128;
      const int r   = idx >> 3;
      const int c8  = (idx & 7) << 3;
      async_ld16(Km + (rowbase + kb * 64 + r) * D_ + col0 + c8,
                 (unsigned int)(KT_OFF + 144 * r + 2 * c8));
    }
#else
#pragma unroll
    for (int i = 0; i < 16; ++i) {
      const int idx = tid + i * 128;
      const int r   = idx >> 5;
      const int c2  = (idx & 31) << 1;
      const unsigned int d = *(const unsigned int*)(
          Km + (rowbase + kb * 64 + r) * D_ + col0 + c2);
      *(unsigned int*)(&Kt[r * 72 + c2]) = d;
    }
#endif
    // stage V tile transposed [hd][key] (transform -> manual)
#pragma unroll
    for (int i = 0; i < 32; ++i) {
      const int idx = tid + i * 128;
      const int r   = idx >> 6;            // key
      const int c   = idx & 63;            // hd
      Vt[c * 72 + r] = Vm[(rowbase + kb * 64 + r) * D_ + col0 + c];
    }
#if ASYNC_LDS
    async_wait0();
#endif
    __syncthreads();

    // scores S = Q K^T * 1/sqrt(HD), causal-masked into LDS
    const v16bf qa0 = load_a_frag(Qt, w * 16, 72, 0);
    const v16bf qa1 = load_a_frag(Qt, w * 16, 72, 32);
#pragma unroll
    for (int nf = 0; nf < 4; ++nf) {
      v8f c = zero8();
      c = wmma_bf16(qa0, load_b_frag(Kt, nf * 16, 72, 0),  c);
      c = wmma_bf16(qa1, load_b_frag(Kt, nf * 16, 72, 32), c);
#pragma unroll
      for (int r = 0; r < 8; ++r) {
        const int lrow = w * 16 + r + 8 * lh;
        const int gq = qb * 64 + lrow;
        const int gk = kb * 64 + nf * 16 + ln;
        sc[lrow * 68 + nf * 16 + ln] = (gk > gq) ? -1e9f : c[r] * 0.125f;
      }
    }
    __syncthreads();

    // online softmax: lanes 0..15 of each wave own one q row each
    if (l < 16) {
      const int lrow = w * 16 + l;
      float mx = m_run;
      for (int j = 0; j < 64; ++j) mx = fmaxf(mx, sc[lrow * 68 + j]);
      const float al = __expf(m_run - mx);
      float ssum = 0.0f;
      for (int j = 0; j < 64; ++j) {
        const float p = __expf(sc[lrow * 68 + j] - mx);
        ssum += p;
        Pt[lrow * 72 + j] = f2b(p);
      }
      l_run = al * l_run + ssum;
      m_run = mx;
      alf[lrow] = al;
    }
    __syncthreads();

    // O = alpha*O + P V
    const v16bf pa0 = load_a_frag(Pt, w * 16, 72, 0);
    const v16bf pa1 = load_a_frag(Pt, w * 16, 72, 32);
#pragma unroll
    for (int nf = 0; nf < 4; ++nf) {
#pragma unroll
      for (int r = 0; r < 8; ++r) o[nf][r] *= alf[w * 16 + r + 8 * lh];
      o[nf] = wmma_bf16(pa0, load_b_frag(Vt, nf * 16, 72, 0),  o[nf]);
      o[nf] = wmma_bf16(pa1, load_b_frag(Vt, nf * 16, 72, 32), o[nf]);
    }
  }

  if (l < 16) linv[w * 16 + l] = 1.0f / l_run;
  __syncthreads();

#pragma unroll
  for (int nf = 0; nf < 4; ++nf)
#pragma unroll
    for (int r = 0; r < 8; ++r) {
      const int lrow = w * 16 + r + 8 * lh;
      const float v = o[nf][r] * linv[lrow];
      Ctx[(rowbase + qb * 64 + lrow) * D_ + col0 + nf * 16 + ln] = f2b(v);
    }
}

// ---------------------------------------------------------------------------
// Kernel 4: x = LayerNorm(x + attn_out); emits f32 + bf16
// ---------------------------------------------------------------------------
__global__ __launch_bounds__(256) void residual_ln(
    const float* __restrict__ xin, const float* __restrict__ ao,
    const float* __restrict__ g, const float* __restrict__ bta,
    float* __restrict__ xout, unsigned short* __restrict__ xb) {
  __shared__ float red[256];
  const int row = blockIdx.x;
  const int tid = threadIdx.x;
  float v[4];
  float s = 0.0f;
#pragma unroll
  for (int i = 0; i < 4; ++i) {
    const int d = tid + i * 256;
    v[i] = xin[row * D_ + d] + ao[row * D_ + d];
    s += v[i];
  }
  red[tid] = s;
  __syncthreads();
  for (int st = 128; st > 0; st >>= 1) {
    if (tid < st) red[tid] += red[tid + st];
    __syncthreads();
  }
  const float mu = red[0] * (1.0f / D_);
  __syncthreads();
  s = 0.0f;
#pragma unroll
  for (int i = 0; i < 4; ++i) { const float t = v[i] - mu; s += t * t; }
  red[tid] = s;
  __syncthreads();
  for (int st = 128; st > 0; st >>= 1) {
    if (tid < st) red[tid] += red[tid + st];
    __syncthreads();
  }
  const float rs = rsqrtf(red[0] * (1.0f / D_) + 1e-5f);
#pragma unroll
  for (int i = 0; i < 4; ++i) {
    const int d = tid + i * 256;
    const float y = (v[i] - mu) * rs * g[d] + bta[d];
    xout[row * D_ + d] = y;
    xb [row * D_ + d] = f2b(y);
  }
}

// ---------------------------------------------------------------------------
// Host launcher
// ---------------------------------------------------------------------------
extern "C" void kernel_launch(void* const* d_in, const int* in_sizes, int n_in,
                              void* d_out, int out_size, void* d_ws,
                              size_t ws_size, hipStream_t stream) {
  (void)in_sizes; (void)n_in; (void)out_size; (void)ws_size;

  const int*   tokens = (const int*)  d_in[0];
  const float* emb    = (const float*)d_in[1];
  const float* qw     = (const float*)d_in[2];
  const float* qbv    = (const float*)d_in[3];
  const float* kw     = (const float*)d_in[4];
  const float* kbv    = (const float*)d_in[5];
  const float* vw     = (const float*)d_in[6];
  const float* vbv    = (const float*)d_in[7];
  const float* ow     = (const float*)d_in[8];
  const float* obv    = (const float*)d_in[9];
  const float* lng    = (const float*)d_in[10];
  const float* lnb    = (const float*)d_in[11];
  const float* outw   = (const float*)d_in[12];
  const float* outb   = (const float*)d_in[13];
  float* out = (float*)d_out;

  // workspace carve-up (~36 MB)
  char* ws = (char*)d_ws;
  size_t off = 0;
  auto carve = [&](size_t bytes) -> void* {
    void* p = ws + off;
    off += (bytes + 255) & ~(size_t)255;
    return p;
  };
  float*          x    = (float*)         carve((size_t)MTOT * D_ * 4);
  unsigned short* xb   = (unsigned short*)carve((size_t)MTOT * D_ * 2);
  unsigned short* qbuf = (unsigned short*)carve((size_t)MTOT * D_ * 2);
  unsigned short* kbuf = (unsigned short*)carve((size_t)MTOT * D_ * 2);
  unsigned short* vbuf = (unsigned short*)carve((size_t)MTOT * D_ * 2);
  unsigned short* ctxb = (unsigned short*)carve((size_t)MTOT * D_ * 2);
  float*          ao   = (float*)         carve((size_t)MTOT * D_ * 4);

  embed_pos<<<MTOT, 256, 0, stream>>>(tokens, emb, x, xb);

  const dim3 gproj(D_ / GBN, MTOT / GBM);      // 8 x 16
  for (int l = 0; l < L_; ++l) {
    const size_t wo = (size_t)l * D_ * D_;
    const size_t bo = (size_t)l * D_;
    gemm_xw<D_, D_><<<gproj, 256, GEMM_LDS_BYTES, stream>>>(
        xb, qw + wo, qbv + bo, nullptr, qbuf);
    gemm_xw<D_, D_><<<gproj, 256, GEMM_LDS_BYTES, stream>>>(
        xb, kw + wo, kbv + bo, nullptr, kbuf);
    gemm_xw<D_, D_><<<gproj, 256, GEMM_LDS_BYTES, stream>>>(
        xb, vw + wo, vbv + bo, nullptr, vbuf);
    attn<<<dim3(S_ / 64, H_, B_), 128, ATTN_LDS_BYTES, stream>>>(
        qbuf, kbuf, vbuf, ctxb);
    gemm_xw<D_, D_><<<gproj, 256, GEMM_LDS_BYTES, stream>>>(
        ctxb, ow + wo, obv + bo, ao, nullptr);
    residual_ln<<<MTOT, 256, 0, stream>>>(x, ao, lng + bo, lnb + bo, x, xb);
  }

  // logits: [2048,1024] x [1024,32000] -> d_out
  gemm_xw<V_, D_><<<dim3(V_ / GBN, MTOT / GBM), 256, GEMM_LDS_BYTES, stream>>>(
      xb, outw, outb, out, nullptr);
}
